// MultiHeadAttention_56203942035766
// MI455X (gfx1250) — matrix operand order repaired and checked by reading, not compiled, not run on hardware
//
#include <hip/hip_runtime.h>
#include <hip/hip_bf16.h>

// ---------------------------------------------------------------------------
// MI455X (gfx1250) multi-head attention with RoPE.
// - All matmuls via v_wmma_f32_16x16x32_bf16 (wave32, 16x16 tiles, K=32).
// - Projection GEMMs: A/B k-slices staged to LDS with
//   global_load_async_to_lds_b128 (ASYNCcnt), double-buffered so the async
//   DMA of slice k+1 overlaps the WMMAs of slice k.
// ---------------------------------------------------------------------------

#define D_MODEL   1024
#define NUM_HEADS 16
#define D_K       64
#define SEQ       2048
#define BATCH     2
#define M_TOTAL   (BATCH * SEQ)   // 4096 rows

typedef __attribute__((ext_vector_type(16))) __bf16 v16bf;
typedef __attribute__((ext_vector_type(8)))  __bf16 v8bf;
typedef __attribute__((ext_vector_type(8)))  float  v8f;

union AFrag { v16bf v; v8bf h[2]; };

static __device__ __forceinline__ v8f wmma_bf16(v16bf a, v16bf b, v8f c) {
  // 8 args: (neg_a, A, neg_b, B, c_mod, C, reuse_a, reuse_b)
  return __builtin_amdgcn_wmma_f32_16x16x32_bf16(false, a, false, b, (short)0, c,
                                                 false, false);
}

// Async DMA: 16 bytes per lane, global -> LDS, tracked by ASYNCcnt.
static __device__ __forceinline__ void async_load_b128(unsigned lds_off,
                                                       const void* gptr) {
  asm volatile("global_load_async_to_lds_b128 %0, %1, off"
               :: "v"(lds_off), "v"((unsigned long long)(size_t)gptr)
               : "memory");
}
static __device__ __forceinline__ void wait_async0() {
  asm volatile("s_wait_asynccnt 0x0" ::: "memory");
}

// ---------------------------------------------------------------------------
// Kernel 1a: fp32 -> bf16 elementwise (for x)
// ---------------------------------------------------------------------------
__global__ void cvt_f32_bf16(const float* __restrict__ in,
                             __bf16* __restrict__ out, int n) {
  int i = blockIdx.x * blockDim.x + threadIdx.x;
  if (i < n) out[i] = (__bf16)in[i];
}

// ---------------------------------------------------------------------------
// Kernel 1b: fp32 W[k][n] -> bf16 WT[n][k]  (1024x1024)
// Transposed weights make the WMMA B-fragment a contiguous K run.
// ---------------------------------------------------------------------------
__global__ void transpose_w(const float* __restrict__ W,
                            __bf16* __restrict__ WT) {
  int i = blockIdx.x * blockDim.x + threadIdx.x;  // over 1024*1024
  int n = i & (D_MODEL - 1);
  int k = i >> 10;
  WT[n * D_MODEL + k] = (__bf16)W[k * D_MODEL + n];
}

// ---------------------------------------------------------------------------
// Kernel 2: C = A(bf16, 4096x1024) x WT(bf16, transposed) + bias
// Block = 8 waves (256 thr), tile 64(M) x 128(N), K-step 32.
// Per k-step the block async-DMAs the A-slice (64x32, 4KB) and B-slice
// (128x32, 8KB) into LDS, double-buffered: 3 uniform b128 chunks per thread
// (no divergent EXEC juggling). Wave tile 16x64 -> 4 WMMAs per k-step.
// Epilogues:
//   mode 0: bias + RoPE -> bf16 [B,H,S,64]    (Q and K)
//   mode 1: bias        -> bf16 [B,H,64,S]    (V^T)
//   mode 2: bias        -> f32 [4096,1024]    (final output)
// ---------------------------------------------------------------------------
__global__ void gemm_bf16(const __bf16* __restrict__ A,
                          const __bf16* __restrict__ WT,
                          const float* __restrict__ bias,
                          __bf16* __restrict__ outBF,
                          float* __restrict__ outF,
                          int mode) {
  __shared__ __attribute__((aligned(64))) __bf16 Abuf[2][64 * 32];
  __shared__ __attribute__((aligned(64))) __bf16 Bbuf[2][128 * 32];

  const int tid  = threadIdx.x;
  const int lane = tid & 31;
  const int wid  = tid >> 5;
  const int wm   = wid & 3;            // 0..3 (M)
  const int wn   = wid >> 2;           // 0..1 (N)
  const int rowblk = blockIdx.x * 64;
  const int colblk = blockIdx.y * 128;
  const int l15 = lane & 15;
  const int hi  = lane >> 4;

  // Stage one 32-wide k-slice (uniform control flow; 3 async b128 per thread):
  //   thread -> A row (tid>>2), B rows (tid>>2) and (tid>>2)+64, seg = tid&3.
  const int srow = tid >> 2;
  const int sseg = (tid & 3) * 8;      // 8-element (16B) group within k-slice
  auto stage = [&](int buf, int kk) {
    async_load_b128((unsigned)(size_t)&Abuf[buf][srow * 32 + sseg],
                    A + (size_t)(rowblk + srow) * D_MODEL + kk + sseg);
    async_load_b128((unsigned)(size_t)&Bbuf[buf][srow * 32 + sseg],
                    WT + (size_t)(colblk + srow) * D_MODEL + kk + sseg);
    async_load_b128((unsigned)(size_t)&Bbuf[buf][(srow + 64) * 32 + sseg],
                    WT + (size_t)(colblk + srow + 64) * D_MODEL + kk + sseg);
  };

  v8f acc[4] = {};

  stage(0, 0);
  wait_async0();
  __syncthreads();

  for (int kk = 0; kk < D_MODEL; kk += 32) {
    const int buf = (kk >> 5) & 1;
    if (kk + 32 < D_MODEL) stage(buf ^ 1, kk + 32);   // overlaps WMMAs below

    const __bf16* ab = &Abuf[buf][0];
    AFrag a;
    a.h[0] = *(const v8bf*)(ab + (wm * 16 + l15) * 32 + 16 * hi);
    a.h[1] = *(const v8bf*)(ab + (wm * 16 + l15) * 32 + 8 + 16 * hi);
    const __bf16* bb = &Bbuf[buf][0];
    #pragma unroll
    for (int t = 0; t < 4; ++t) {
      v16bf b = *(const v16bf*)(bb + (wn * 64 + t * 16 + l15) * 32 + 16 * hi);
      acc[t] = wmma_bf16(a.v, b, acc[t]);
    }

    wait_async0();        // own next-slice DMAs done
    __syncthreads();      // everyone's done; reads of `buf` finished
  }

  // Epilogue. C layout: elem j of frag t -> (m = row0 + j + 8*hi,
  //                                          n = col0 + t*16 + l15)
  const int row0 = rowblk + wm * 16;
  const int col0 = colblk + wn * 64;
  #pragma unroll
  for (int t = 0; t < 4; ++t) {
    const int n  = col0 + t * 16 + l15;
    const float bn = bias[n];
    #pragma unroll
    for (int j = 0; j < 8; ++j) {
      const int m = row0 + j + 8 * hi;
      float v = acc[t][j] + bn;
      if (mode == 2) {
        outF[(size_t)m * D_MODEL + n] = v;
      } else {
        const int b = m >> 11;            // SEQ = 2048
        const int s = m & (SEQ - 1);
        const int h = n >> 6;             // D_K = 64
        const int d = n & 63;
        if (mode == 0) {
          // RoPE: partner element (d^1) lives in lane^1 of this fragment.
          float other = __shfl_xor(v, 1, 32);
          float ang = (float)s *
                      __expf(-(float)(d & ~1) * (9.210340371976184f / 64.f));
          float c, sn;
          __sincosf(ang, &sn, &c);
          float r = v * c + ((d & 1) ? other * sn : -other * sn);
          outBF[(((size_t)(b * NUM_HEADS + h) * SEQ + s) * D_K) + d] = (__bf16)r;
        } else {  // mode 1: V^T  [B,H,64,S]
          outBF[(((size_t)(b * NUM_HEADS + h) * D_K + d) * SEQ) + s] = (__bf16)v;
        }
      }
    }
  }
}

// ---------------------------------------------------------------------------
// Kernel 3: causal flash attention per (b,h).
// Block = 4 waves (128 threads); wave w owns q rows [q0, q0+16).
// Streams 32-key tiles: QK^T = 4 WMMAs, online softmax, P staged through LDS
// (C-layout -> A-layout conversion), PV = 4 WMMAs.
// ---------------------------------------------------------------------------
__global__ void attention_kernel(const __bf16* __restrict__ QR,
                                 const __bf16* __restrict__ KR,
                                 const __bf16* __restrict__ VT,
                                 __bf16* __restrict__ AT) {
  __shared__ __attribute__((aligned(32))) __bf16 pbuf[4][16 * 40];  // pad 40

  const int lane = threadIdx.x & 31;
  const int w    = threadIdx.x >> 5;     // wave 0..3
  const int l15  = lane & 15;
  const int hi   = lane >> 4;
  const int bh   = blockIdx.y;           // 0..31
  const int b    = bh >> 4;
  const int h    = bh & 15;
  const int q0   = blockIdx.x * 64 + w * 16;

  const __bf16* Q = QR + (size_t)bh * SEQ * D_K;
  const __bf16* K = KR + (size_t)bh * SEQ * D_K;
  const __bf16* V = VT + (size_t)bh * D_K * SEQ;   // [64][S]

  // Q fragment: 16 rows x dk=64 -> two 16x32 A-frags, loaded once.
  AFrag aq[2];
  {
    const __bf16* qrow = Q + (size_t)(q0 + l15) * D_K;
    #pragma unroll
    for (int c = 0; c < 2; ++c) {
      aq[c].h[0] = *(const v8bf*)(qrow + c * 32 + 16 * hi);
      aq[c].h[1] = *(const v8bf*)(qrow + c * 32 + 8 + 16 * hi);
    }
  }

  float rmax[8], rsum[8];
  v8f o[4] = {};
  #pragma unroll
  for (int j = 0; j < 8; ++j) { rmax[j] = -1e30f; rsum[j] = 0.f; }

  __bf16* pb = &pbuf[w][0];

  // kt < q0+16 guarantees kt <= q for every owned row q (q0 multiple of 16,
  // kt multiple of 32) -> every row has >=1 unmasked key per tile.
  for (int kt = 0; kt < q0 + 16; kt += 32) {
    // ---- scores S = Q K^T for 32 keys (two 16-key N tiles) ----
    v8f s[2];
    #pragma unroll
    for (int nt = 0; nt < 2; ++nt) {
      const __bf16* krow = K + (size_t)(kt + nt * 16 + l15) * D_K;
      __builtin_prefetch(krow + 32 * D_K, 0, 3);         // next key tile
      v16bf b0 = *(const v16bf*)(krow + 16 * hi);        // dk 0..31
      v16bf b1 = *(const v16bf*)(krow + 32 + 16 * hi);   // dk 32..63
      v8f z = {};
      z = wmma_bf16(aq[0].v, b0, z);
      z = wmma_bf16(aq[1].v, b1, z);
      s[nt] = z;
    }

    // ---- scale, causal mask, online softmax, P -> LDS (bf16) ----
    float alpha[8];
    #pragma unroll
    for (int j = 0; j < 8; ++j) {
      const int qg = q0 + j + 8 * hi;
      float v0 = s[0][j] * 0.125f;          // 1/sqrt(64)
      float v1 = s[1][j] * 0.125f;
      if (kt + l15      > qg) v0 = -1e30f;
      if (kt + 16 + l15 > qg) v1 = -1e30f;
      float rm = fmaxf(v0, v1);
      #pragma unroll
      for (int x = 1; x < 16; x <<= 1) rm = fmaxf(rm, __shfl_xor(rm, x, 32));
      const float nm = fmaxf(rmax[j], rm);
      const float al = __expf(rmax[j] - nm);
      const float p0 = __expf(v0 - nm);
      const float p1 = __expf(v1 - nm);
      float ps = p0 + p1;
      #pragma unroll
      for (int x = 1; x < 16; x <<= 1) ps += __shfl_xor(ps, x, 32);
      rsum[j] = rsum[j] * al + ps;
      rmax[j] = nm;
      alpha[j] = al;
      const int m = j + 8 * hi;
      pb[m * 40 + l15]      = (__bf16)p0;
      pb[m * 40 + 16 + l15] = (__bf16)p1;
    }
    asm volatile("s_wait_dscnt 0x0" ::: "memory");  // lds stores -> lds loads

    // ---- reload P in A-frag layout (16x32 bf16) ----
    AFrag pa;
    pa.h[0] = *(const v8bf*)(pb + l15 * 40 + 16 * hi);
    pa.h[1] = *(const v8bf*)(pb + l15 * 40 + 8 + 16 * hi);

    // ---- O = diag(alpha) O + P V ----
    #pragma unroll
    for (int t = 0; t < 4; ++t) {
      #pragma unroll
      for (int j = 0; j < 8; ++j) o[t][j] *= alpha[j];
      const __bf16* vrow = V + (size_t)(t * 16 + l15) * SEQ + kt;
      v16bf vb = *(const v16bf*)(vrow + 16 * hi);  // keys contiguous in V^T
      o[t] = wmma_bf16(pa.v, vb, o[t]);
    }
  }

  // ---- normalize, store bf16 to attention buffer [4096, 1024] ----
  #pragma unroll
  for (int t = 0; t < 4; ++t) {
    #pragma unroll
    for (int j = 0; j < 8; ++j) {
      const int sg  = q0 + j + 8 * hi;
      const int col = h * D_K + t * 16 + l15;
      AT[(size_t)(b * SEQ + sg) * D_MODEL + col] = (__bf16)(o[t][j] / rsum[j]);
    }
  }
}

// ---------------------------------------------------------------------------
// Launch
// ---------------------------------------------------------------------------
extern "C" void kernel_launch(void* const* d_in, const int* in_sizes, int n_in,
                              void* d_out, int out_size, void* d_ws, size_t ws_size,
                              hipStream_t stream) {
  const float* x  = (const float*)d_in[0];
  const float* Wq = (const float*)d_in[1];
  const float* bq = (const float*)d_in[2];
  const float* Wk = (const float*)d_in[3];
  const float* bk = (const float*)d_in[4];
  const float* Wv = (const float*)d_in[5];
  const float* bv = (const float*)d_in[6];
  const float* Wo = (const float*)d_in[7];
  const float* bo = (const float*)d_in[8];

  const size_t NX = (size_t)M_TOTAL * D_MODEL;   // 4,194,304 elements
  const size_t NW = (size_t)D_MODEL * D_MODEL;   // 1,048,576 elements

  char* ws = (char*)d_ws;
  __bf16* XB  = (__bf16*)(ws);
  __bf16* WQT = (__bf16*)(ws + 2 * (NX));
  __bf16* WKT = (__bf16*)(ws + 2 * (NX + NW));
  __bf16* WVT = (__bf16*)(ws + 2 * (NX + 2 * NW));
  __bf16* WOT = (__bf16*)(ws + 2 * (NX + 3 * NW));
  __bf16* QR  = (__bf16*)(ws + 2 * (NX + 4 * NW));
  __bf16* KR  = (__bf16*)(ws + 2 * (2 * NX + 4 * NW));
  __bf16* VTt = (__bf16*)(ws + 2 * (3 * NX + 4 * NW));
  __bf16* AT  = (__bf16*)(ws + 2 * (4 * NX + 4 * NW));

  cvt_f32_bf16<<<dim3((unsigned)(NX / 256)), dim3(256), 0, stream>>>(x, XB, (int)NX);
  transpose_w<<<dim3((unsigned)(NW / 256)), dim3(256), 0, stream>>>(Wq, WQT);
  transpose_w<<<dim3((unsigned)(NW / 256)), dim3(256), 0, stream>>>(Wk, WKT);
  transpose_w<<<dim3((unsigned)(NW / 256)), dim3(256), 0, stream>>>(Wv, WVT);
  transpose_w<<<dim3((unsigned)(NW / 256)), dim3(256), 0, stream>>>(Wo, WOT);

  dim3 gg(M_TOTAL / 64, D_MODEL / 128);
  gemm_bf16<<<gg, 256, 0, stream>>>(XB, WQT, bq, QR,  nullptr, 0);  // Q + RoPE
  gemm_bf16<<<gg, 256, 0, stream>>>(XB, WKT, bk, KR,  nullptr, 0);  // K + RoPE
  gemm_bf16<<<gg, 256, 0, stream>>>(XB, WVT, bv, VTt, nullptr, 1);  // V^T

  attention_kernel<<<dim3(SEQ / 64, BATCH * NUM_HEADS), 128, 0, stream>>>(
      QR, KR, VTt, AT);

  gemm_bf16<<<gg, 256, 0, stream>>>(AT, WOT, bo, nullptr, (float*)d_out, 2);
}
